// ArtistBERT_81389630259578
// MI455X (gfx1250) — matrix operand order repaired and checked by reading, not compiled, not run on hardware
//
#include <hip/hip_runtime.h>

typedef __attribute__((ext_vector_type(16))) _Float16 v16h;
typedef __attribute__((ext_vector_type(8)))  _Float16 v8h;
typedef __attribute__((ext_vector_type(8)))  float    v8f;
typedef __attribute__((ext_vector_type(4)))  float    v4f;
typedef __attribute__((ext_vector_type(4)))  int      v4i;
typedef __attribute__((ext_vector_type(8)))  int      v8i;

#define B_TOT   16384
#define S_LEN   17
#define V_DIM   29
#define D_DIM   128
#define NSEG    4
#define NLAYER  6
#define EPSF    1e-5f

#define NB       4              // batches per block
#define RP       32             // padded rows per batch (S 17 -> 32)
#define ROWS     (NB * RP)      // 128 padded rows per block
#define NTHREADS 256            // 8 wave32; wave w owns rows [16w, 16w+16)

// ---- dynamic LDS layout (bytes) ----
#define OFF_XS   0                       // float [ROWS][D]      65536  fp32 residual
#define OFF_WSW  65536                   // f16, swizzled weight 32768  (B-fragment layout)
#define OFF_QH   (OFF_WSW + 32768)       // f16  [ROWS][D]       32768
#define OFF_KH   (OFF_QH + 32768)        // f16  [ROWS][D]       32768
#define OFF_VT   (OFF_KH + 32768)        // f16  [D][ROWS]       32768  (v transposed)
#define OFF_SC   (OFF_VT + 32768)        // float[NB][RP][RP]    16384  (scores / emb tbl)
#define OFF_SA   (OFF_SC + 16384)        // f16  [NB][RP][RP]     8192  (softmax A / misc)
#define OFF_WF32 (OFF_SA + 8192)         // float[D][D]          65536  TDM fp32 staging
#define SMEM_BYTES (OFF_WF32 + 65536)    // 286720 B = 280 KB  (< 320 KB WGP LDS)

#define WMMA_F16(a, b, c) \
  __builtin_amdgcn_wmma_f32_16x16x32_f16(false, (a), false, (b), (short)0, (c), false, false)

// ---------------- Tensor Data Mover: async 2D fp32 tile -> LDS ----------------
// D# per CDNA5 ISA Sec 8.3/8.4: group0 = {count|flags, lds_addr,
// global_addr[31:0], global_addr[56:32]|type=2}; group1 packs data_size=4B,
// tensor_dim0/1 = tile_dim0/1 = dims, tensor_dim0_stride = row stride.
__device__ inline void tdm_load_2d(unsigned lds_off, const float* gptr,
                                   unsigned d0, unsigned d1) {
  unsigned long long ga = (unsigned long long)gptr;
  v4i g0;
  g0[0] = 1;                                        // count=1 (valid user descriptor)
  g0[1] = (int)lds_off;                             // lds_addr (bytes)
  g0[2] = (int)(unsigned)(ga & 0xffffffffu);        // global_addr[31:0]
  g0[3] = (int)((unsigned)((ga >> 32) & 0x1ffffffu) | (2u << 30)); // [56:32] | type=2
  v8i g1;
  g1[0] = (int)(2u << 16);                          // data_size=2 -> 4 bytes
  g1[1] = (int)(d0 << 16);                          // tensor_dim0[15:0]
  g1[2] = (int)(d1 << 16);                          // tensor_dim0 hi=0 | tensor_dim1 lo
  g1[3] = (int)(d0 << 16);                          // tensor_dim1 hi=0 | tile_dim0
  g1[4] = (int)d1;                                  // tile_dim1 | tile_dim2=0
  g1[5] = (int)d0;                                  // tensor_dim0_stride lo32
  g1[6] = 0;                                        // stride hi | tensor_dim1_stride lo
  g1[7] = 0;
  asm volatile("tensor_load_to_lds %0, %1" :: "s"(g0), "s"(g1) : "memory");
}

// ---------------- fragment loaders (ISA 7.12.2, wave32) ----------------
// A 16x32 f16 from row-major [rows][stride]: two contiguous 16B chunks per lane.
__device__ inline v16h load_a_h(const _Float16* base, int stride, int row0, int k0, int lane) {
  const int m = lane & 15, grp = lane >> 4;
  const _Float16* p = base + (row0 + m) * stride + k0 + grp * 16;
  const v8h lo = *(const v8h*)p;
  const v8h hi = *(const v8h*)(p + 8);
  v16h a;
#pragma unroll
  for (int j = 0; j < 8; ++j) { a[j] = lo[j]; a[8 + j] = hi[j]; }
  return a;
}
// A 16x32 from fp32 row-major LDS, converted on the fly (4x v4f vector loads).
__device__ inline v16h load_a_f(const float* base, int stride, int row0, int k0, int lane) {
  const int m = lane & 15, grp = lane >> 4;
  const float* p = base + (row0 + m) * stride + k0 + grp * 16;
  const v4f x0 = *(const v4f*)p, x1 = *(const v4f*)(p + 4);
  const v4f y0 = *(const v4f*)(p + 8), y1 = *(const v4f*)(p + 12);
  v16h a;
#pragma unroll
  for (int j = 0; j < 4; ++j) {
    a[j] = (_Float16)x0[j]; a[4 + j]  = (_Float16)x1[j];
    a[8 + j] = (_Float16)y0[j]; a[12 + j] = (_Float16)y1[j];
  }
  return a;
}
// B 32x16 from swizzled weight buffer: one contiguous 32B block per lane.
__device__ inline v16h load_b_sw(const _Float16* wsw, int frag, int lane) {
  return *(const v16h*)(wsw + ((frag * 32 + lane) << 4));
}
// B 32x16 with B[k][n] = src[(nbase+n)*stride + k0+k]  (contiguous; for Q@K^T and attn@V^T)
__device__ inline v16h load_b_rows(const _Float16* src, int stride, int nbase, int k0, int lane) {
  const int n = lane & 15, grp = lane >> 4;
  return *(const v16h*)(src + (nbase + n) * stride + k0 + grp * 16);
}

// swizzle fp32 LDS staging -> f16 B-fragment layout (frag = kt*8+nt)
__device__ inline void swizzle_wf32(const float* wf32, _Float16* wsw, int tid) {
  for (int idx = tid; idx < D_DIM * D_DIM; idx += NTHREADS) {
    const int k = idx >> 7, n = idx & 127;
    const int kt = k >> 5, kr = k & 31, g = kr >> 4, j = kr & 15;
    const int nt = n >> 4, nc = n & 15;
    wsw[(((kt * 8 + nt) * 32) + (g * 16 + nc)) * 16 + j] = (_Float16)wf32[idx];
  }
}

// pipeline: consume TDM-staged weight (wf32 -> wsw), then issue TDM for next one
__device__ inline void pipeline_step(const float* wf32, _Float16* wsw,
                                     const float* nextW, int tid, int wave) {
  if (wave == 0) __builtin_amdgcn_s_wait_tensorcnt(0);
  __syncthreads();                       // TDM data visible; prev GEMM done (wsw free)
  swizzle_wf32(wf32, wsw, tid);
  __syncthreads();                       // wsw ready; wf32 free for next TDM
  if (wave == 0 && nextW) tdm_load_2d(OFF_WF32, nextW, D_DIM, D_DIM);
}

// dst[rows16 x D] = relu(A @ Wsw + bias), f16 row-major store
__device__ inline void gemm_to_h(const v16h* ax, const _Float16* wsw, _Float16* dst,
                                 const float* bias, int row0, int grp, int ncol, int lane) {
#pragma unroll
  for (int nt = 0; nt < 8; ++nt) {
    v8f acc = {};
#pragma unroll
    for (int kt = 0; kt < 4; ++kt)
      acc = WMMA_F16(ax[kt], load_b_sw(wsw, kt * 8 + nt, lane), acc);
    const int col = nt * 16 + ncol;
    const float bv = bias[col];
#pragma unroll
    for (int i = 0; i < 8; ++i)
      dst[(row0 + i + 8 * grp) * D_DIM + col] = (_Float16)fmaxf(acc[i] + bv, 0.f);
  }
}
// same but transposed store: dst[col][row]  (for V so attn B-fragments are contiguous)
__device__ inline void gemm_to_hT(const v16h* ax, const _Float16* wsw, _Float16* dst,
                                  const float* bias, int row0, int grp, int ncol, int lane) {
#pragma unroll
  for (int nt = 0; nt < 8; ++nt) {
    v8f acc = {};
#pragma unroll
    for (int kt = 0; kt < 4; ++kt)
      acc = WMMA_F16(ax[kt], load_b_sw(wsw, kt * 8 + nt, lane), acc);
    const int col = nt * 16 + ncol;
    const float bv = bias[col];
#pragma unroll
    for (int i = 0; i < 8; ++i)
      dst[col * ROWS + (row0 + i + 8 * grp)] = (_Float16)fmaxf(acc[i] + bv, 0.f);
  }
}

__device__ inline void layernorm_rows(float* xs, const float* g, const float* b, int tid) {
  if (tid < ROWS) {
    float* row = xs + tid * D_DIM;
    float mu = 0.f;
#pragma unroll 8
    for (int d = 0; d < D_DIM; ++d) mu += row[d];
    mu *= (1.f / D_DIM);
    float var = 0.f;
#pragma unroll 8
    for (int d = 0; d < D_DIM; ++d) { float t = row[d] - mu; var += t * t; }
    var *= (1.f / D_DIM);
    const float inv = rsqrtf(var + EPSF);
#pragma unroll 8
    for (int d = 0; d < D_DIM; ++d) row[d] = (row[d] - mu) * inv * g[d] + b[d];
  }
}

__global__ __launch_bounds__(NTHREADS)
void artistbert_kernel(const float* __restrict__ X,    const float* __restrict__ mask_in,
                       const float* __restrict__ seg_in,
                       const float* __restrict__ We,   const float* __restrict__ be,
                       const float* __restrict__ Wp,   const float* __restrict__ bp,
                       const float* __restrict__ Wsg,  const float* __restrict__ bsg,
                       const float* __restrict__ g1,   const float* __restrict__ b1,
                       const float* __restrict__ Wq,   const float* __restrict__ bq,
                       const float* __restrict__ Wk,   const float* __restrict__ bk,
                       const float* __restrict__ Wv,   const float* __restrict__ bv,
                       const float* __restrict__ g2,   const float* __restrict__ b2,
                       const float* __restrict__ Wd,   const float* __restrict__ bd,
                       const float* __restrict__ Wo,   const float* __restrict__ bo,
                       float* __restrict__ out) {
  extern __shared__ char smem[];
  float*    xs   = (float*)   (smem + OFF_XS);
  _Float16* wsw  = (_Float16*)(smem + OFF_WSW);
  _Float16* qh   = (_Float16*)(smem + OFF_QH);
  _Float16* kh   = (_Float16*)(smem + OFF_KH);
  _Float16* vt   = (_Float16*)(smem + OFF_VT);
  float*    sc   = (float*)   (smem + OFF_SC);
  _Float16* sa   = (_Float16*)(smem + OFF_SA);
  float*    wf32 = (float*)   (smem + OFF_WF32);

  const int tid   = threadIdx.x;
  const int lane  = tid & 31;
  const int wave  = tid >> 5;
  const int grp   = lane >> 4;
  const int ncol  = lane & 15;
  const int row0  = wave * 16;
  const int bl    = wave >> 1;        // local batch for attention (2 waves / batch)
  const int gbase = blockIdx.x * NB;

  // prefetch layer-0 Wq via Tensor Data Mover while we do the embedding
  if (wave == 0) tdm_load_2d(OFF_WF32, Wq, D_DIM, D_DIM);

  // ============ embedding: x = X@We + be + (Wp+bp) + seg@Wsg + bsg ============
  // stage X (K padded 29->32) into qh as [ROWS][32] f16
  for (int idx = tid; idx < ROWS * 32; idx += NTHREADS) {
    const int r = idx >> 5, c = idx & 31;
    const int b = r >> 5, s = r & 31;
    float v = 0.f;
    if (s < S_LEN && c < V_DIM) v = X[((gbase + b) * S_LEN + s) * V_DIM + c];
    qh[idx] = (_Float16)v;
  }
  // stage We (K padded 29->32) into wsw, swizzled (kt=0, frag=nt)
  for (int idx = tid; idx < 32 * D_DIM; idx += NTHREADS) {
    const int k = idx >> 7, n = idx & 127;
    const int g = k >> 4, j = k & 15, nt = n >> 4, nc = n & 15;
    wsw[((nt * 32) + (g * 16 + nc)) * 16 + j] = (_Float16)(k < V_DIM ? We[k * D_DIM + n] : 0.f);
  }
  // stage emb_base[s][col] = be + Wp[s] + bp + bsg into sc (zero for padded s)
  for (int idx = tid; idx < RP * D_DIM; idx += NTHREADS) {
    const int s = idx >> 7, col = idx & 127;
    sc[idx] = (s < S_LEN) ? (be[col] + Wp[s * D_DIM + col] + bp[col] + bsg[col]) : 0.f;
  }
  // stage Wsg as 5 rows (row 4 = zeros) + per-row segment id (4 for padded rows)
  {
    float* wsgl = (float*)sa;                   // 5*128 floats = 2.5 KB
    int*   sidl = (int*)(smem + OFF_SA + 2560); // 512 B
    for (int idx = tid; idx < 5 * D_DIM; idx += NTHREADS)
      wsgl[idx] = (idx < NSEG * D_DIM) ? Wsg[idx] : 0.f;
    if (tid < ROWS) {
      const int s = tid & 31, b = tid >> 5;
      int sid = NSEG;                           // zero row for padded rows
      if (s < S_LEN) {
        sid = 0;
        const float* sg = seg_in + ((gbase + b) * S_LEN + s) * NSEG;
#pragma unroll
        for (int j = 1; j < NSEG; ++j) if (sg[j] > 0.5f) sid = j;
      }
      sidl[tid] = sid;
    }
  }
  __syncthreads();
  {
    const float* wsgl = (const float*)sa;
    const int*   sidl = (const int*)(smem + OFF_SA + 2560);
    const v16h a = load_a_h(qh, 32, row0, 0, lane);   // single K=32 step
    int sidv[8];                                      // hoist per-row segment ids
#pragma unroll
    for (int i = 0; i < 8; ++i) sidv[i] = sidl[row0 + i + 8 * grp] * D_DIM;
#pragma unroll
    for (int nt = 0; nt < 8; ++nt) {
      v8f acc = {};
      acc = WMMA_F16(a, load_b_sw(wsw, nt, lane), acc);
      const int col = nt * 16 + ncol;
#pragma unroll
      for (int i = 0; i < 8; ++i) {
        const int rg = row0 + i + 8 * grp;
        const int s = rg & 31;
        // branchless: padded rows get 0 + 0 + 0 (acc=0, tables zeroed)
        xs[rg * D_DIM + col] = acc[i] + sc[s * D_DIM + col] + wsgl[sidv[i] + col];
      }
    }
  }
  __syncthreads();

  // ============ 6 transformer layers (TDM-pipelined weights) ============
  for (int li = 0; li < NLAYER; ++li) {
    const int off = li * D_DIM * D_DIM;

    // shared A fragments of x for q/k/v (xs constant through all three)
    v16h ax[4];
#pragma unroll
    for (int kt = 0; kt < 4; ++kt) ax[kt] = load_a_f(xs, D_DIM, row0, kt * 32, lane);

    pipeline_step(wf32, wsw, Wk + off, tid, wave);                 // wsw <- Wq[li]
    gemm_to_h(ax, wsw, qh, bq + li * D_DIM, row0, grp, ncol, lane);
    pipeline_step(wf32, wsw, Wv + off, tid, wave);                 // wsw <- Wk[li]
    gemm_to_h(ax, wsw, kh, bk + li * D_DIM, row0, grp, ncol, lane);
    pipeline_step(wf32, wsw, Wd + off, tid, wave);                 // wsw <- Wv[li]
    gemm_to_hT(ax, wsw, vt, bv + li * D_DIM, row0, grp, ncol, lane);
    __syncthreads();

    // ---- scores = q @ k^T  (per batch, padded 32x32, K=128) ----
    {
      v16h aq[4];
#pragma unroll
      for (int kt = 0; kt < 4; ++kt) aq[kt] = load_a_h(qh, D_DIM, row0, kt * 32, lane);
#pragma unroll
      for (int nt2 = 0; nt2 < 2; ++nt2) {
        v8f acc = {};
#pragma unroll
        for (int kt = 0; kt < 4; ++kt)
          acc = WMMA_F16(aq[kt], load_b_rows(kh, D_DIM, bl * RP + nt2 * 16, kt * 32, lane), acc);
#pragma unroll
        for (int i = 0; i < 8; ++i) {
          const int s1 = (row0 & 31) + i + 8 * grp;
          sc[(bl * RP + s1) * RP + nt2 * 16 + ncol] = acc[i];
        }
      }
    }
    __syncthreads();

    // ---- softmax over 17 valid keys -> zero-padded f16 A staging ----
    if (tid < ROWS) {
      const int s1 = tid & 31;
      const float* srow = sc + tid * RP;
      _Float16* drow = sa + tid * RP;
      if (s1 < S_LEN) {
        float mx = srow[0];
#pragma unroll
        for (int j = 1; j < S_LEN; ++j) mx = fmaxf(mx, srow[j]);
        float e[S_LEN], sum = 0.f;
#pragma unroll
        for (int j = 0; j < S_LEN; ++j) { e[j] = __expf(srow[j] - mx); sum += e[j]; }
        const float inv = 1.f / sum;
#pragma unroll
        for (int j = 0; j < S_LEN; ++j) drow[j] = (_Float16)(e[j] * inv);
#pragma unroll
        for (int j = S_LEN; j < RP; ++j) drow[j] = (_Float16)0.f;
      } else {
#pragma unroll
        for (int j = 0; j < RP; ++j) drow[j] = (_Float16)0.f;
      }
    }
    __syncthreads();

    // ---- attn = softmax @ v  (K=32, B from transposed v), masked residual ----
    {
      const v16h a = load_a_h(sa, RP, row0, 0, lane);
      float mrow[8];
#pragma unroll
      for (int i = 0; i < 8; ++i) {
        const int rg = row0 + i + 8 * grp;
        const int s = rg & 31, b_l = rg >> 5;
        mrow[i] = (s < S_LEN) ? mask_in[(gbase + b_l) * S_LEN + s] : 0.f;
      }
#pragma unroll
      for (int nt = 0; nt < 8; ++nt) {
        v8f acc = {};
        acc = WMMA_F16(a, load_b_rows(vt, ROWS, nt * 16, bl * RP, lane), acc);
        const int col = nt * 16 + ncol;
#pragma unroll
        for (int i = 0; i < 8; ++i)
          xs[(row0 + i + 8 * grp) * D_DIM + col] += acc[i] * mrow[i];
      }
    }
    __syncthreads();
    layernorm_rows(xs, g1 + li * D_DIM, b1 + li * D_DIM, tid);
    __syncthreads();

    // ---- dense sublayer: x = LN(x + x@Wd + bd) ----
    pipeline_step(wf32, wsw, (li + 1 < NLAYER) ? (Wq + off + D_DIM * D_DIM) : nullptr,
                  tid, wave);                                      // wsw <- Wd[li]
    v16h ad[4];
#pragma unroll
    for (int kt = 0; kt < 4; ++kt) ad[kt] = load_a_f(xs, D_DIM, row0, kt * 32, lane);
    __syncthreads();
#pragma unroll
    for (int nt = 0; nt < 8; ++nt) {
      v8f acc = {};
#pragma unroll
      for (int kt = 0; kt < 4; ++kt)
        acc = WMMA_F16(ad[kt], load_b_sw(wsw, kt * 8 + nt, lane), acc);
      const int col = nt * 16 + ncol;
      const float bdv = bd[li * D_DIM + col];
#pragma unroll
      for (int i = 0; i < 8; ++i)
        xs[(row0 + i + 8 * grp) * D_DIM + col] += acc[i] + bdv;   // disjoint rows per wave
    }
    __syncthreads();
    layernorm_rows(xs, g2 + li * D_DIM, b2 + li * D_DIM, tid);
    __syncthreads();
  }

  // ============ output: y = x @ Wo + bo  (N padded 29->32, frag = kt*2+nt) ============
  for (int idx = tid; idx < 8 * 512; idx += NTHREADS) {
    const int fid = idx >> 9, rem = idx & 511, ln = rem >> 4, j = rem & 15;
    const int kt = fid >> 1, nt = fid & 1;
    const int k = kt * 32 + (ln >> 4) * 16 + j, n = nt * 16 + (ln & 15);
    wsw[idx] = (_Float16)(n < V_DIM ? Wo[k * V_DIM + n] : 0.f);
  }
  v16h ao[4];
#pragma unroll
  for (int kt = 0; kt < 4; ++kt) ao[kt] = load_a_f(xs, D_DIM, row0, kt * 32, lane);
  __syncthreads();
#pragma unroll
  for (int nt2 = 0; nt2 < 2; ++nt2) {
    v8f acc = {};
#pragma unroll
    for (int kt = 0; kt < 4; ++kt)
      acc = WMMA_F16(ao[kt], load_b_sw(wsw, kt * 2 + nt2, lane), acc);
    const int col = nt2 * 16 + ncol;
#pragma unroll
    for (int i = 0; i < 8; ++i) {
      const int rg = row0 + i + 8 * grp;
      const int b_l = rg >> 5, s = rg & 31;
      if (s < S_LEN && col < V_DIM)
        out[((gbase + b_l) * S_LEN + s) * V_DIM + col] = acc[i] + bo[col];
    }
  }
}

extern "C" void kernel_launch(void* const* d_in, const int* in_sizes, int n_in,
                              void* d_out, int out_size, void* d_ws, size_t ws_size,
                              hipStream_t stream) {
  const float* X    = (const float*)d_in[0];
  const float* mask = (const float*)d_in[1];
  const float* seg  = (const float*)d_in[2];
  const float* We   = (const float*)d_in[3];
  const float* be   = (const float*)d_in[4];
  const float* Wp   = (const float*)d_in[5];
  const float* bp   = (const float*)d_in[6];
  const float* Wsg  = (const float*)d_in[7];
  const float* bsg  = (const float*)d_in[8];
  const float* g1   = (const float*)d_in[9];
  const float* b1   = (const float*)d_in[10];
  const float* Wq   = (const float*)d_in[11];
  const float* bq   = (const float*)d_in[12];
  const float* Wk   = (const float*)d_in[13];
  const float* bk   = (const float*)d_in[14];
  const float* Wv   = (const float*)d_in[15];
  const float* bv   = (const float*)d_in[16];
  const float* g2   = (const float*)d_in[17];
  const float* b2   = (const float*)d_in[18];
  const float* Wd   = (const float*)d_in[19];
  const float* bd   = (const float*)d_in[20];
  const float* Wo   = (const float*)d_in[21];
  const float* bo   = (const float*)d_in[22];
  float* out = (float*)d_out;

  (void)hipFuncSetAttribute((const void*)artistbert_kernel,
                            hipFuncAttributeMaxDynamicSharedMemorySize, SMEM_BYTES);
  dim3 grid(B_TOT / NB), block(NTHREADS);
  artistbert_kernel<<<grid, block, SMEM_BYTES, stream>>>(
      X, mask, seg, We, be, Wp, bp, Wsg, bsg, g1, b1,
      Wq, bq, Wk, bk, Wv, bv, g2, b2, Wd, bd, Wo, bo, out);
}
